// PixelGatingAlign_29609504538961
// MI455X (gfx1250) — matrix operand order repaired and checked by reading, not compiled, not run on hardware
//
#include <hip/hip_runtime.h>
#include <hip/hip_bf16.h>

// ---------------------------------------------------------------------------
// Types for CDNA5 WMMA (wave32)
// ---------------------------------------------------------------------------
typedef __attribute__((ext_vector_type(16))) __bf16 v16bf;
typedef __attribute__((ext_vector_type(8)))  float  v8f;

union Frag {
    uint4 q[2];   // 32 bytes
    v16bf v;      // 16 bf16
};

__device__ __forceinline__ unsigned short f32_to_bf16(float f) {
    unsigned u = __float_as_uint(f);
    unsigned r = (u + 0x7FFFu + ((u >> 16) & 1u)) >> 16;   // round-nearest-even
    return (unsigned short)r;
}

// float <-> order-preserving unsigned (for atomicMax over signed floats)
__device__ __forceinline__ unsigned f2ord(float f) {
    unsigned u = __float_as_uint(f);
    return (u & 0x80000000u) ? ~u : (u | 0x80000000u);
}
__device__ __forceinline__ float ord2f(unsigned o) {
    unsigned u = (o & 0x80000000u) ? (o & 0x7FFFFFFFu) : ~o;
    return __uint_as_float(u);
}

// ---------------------------------------------------------------------------
// Weight conversion: f32 OIHW [outC][Cin][3][3]  ->  bf16 [tap][outC][Cin]
// ---------------------------------------------------------------------------
__global__ void wconv_kernel(const float* __restrict__ src,
                             unsigned short* __restrict__ dst,
                             int outC, int Cin) {
    size_t n = (size_t)9 * outC * Cin;
    size_t d = (size_t)blockIdx.x * blockDim.x + threadIdx.x;
    if (d >= n) return;
    size_t oc_ci = (size_t)outC * Cin;
    int t = (int)(d / oc_ci);
    size_t r = d % oc_ci;
    int o = (int)(r / Cin);
    int i = (int)(r % Cin);
    dst[d] = f32_to_bf16(src[((size_t)o * Cin + i) * 9 + t]);
}

// ---------------------------------------------------------------------------
// Correlation: corr[b,k,y,x] = sum_c key[b,c,y,x] * f1[b,c,y+i-2,x+j-2]
// plus global max (ordered-uint atomicMax) for the 20/corr.max() scale.
// ---------------------------------------------------------------------------
__global__ __launch_bounds__(256)
void corr_kernel(const float* __restrict__ f2, const float* __restrict__ f1,
                 float* __restrict__ corr, unsigned* __restrict__ gmax,
                 int C, int H, int W, int Npix) {
    int p = blockIdx.x * blockDim.x + threadIdx.x;
    int HW = H * W;
    float acc[25];
#pragma unroll
    for (int k = 0; k < 25; ++k) acc[k] = 0.f;
    float m = -1e30f;
    if (p < Npix) {
        int b = p / HW, r = p % HW, y = r / W, x = r % W;
        for (int c = 0; c < C; ++c) {
            size_t base = ((size_t)(b * C + c)) * HW;
            float v2 = f2[base + r];
            const float* src = f1 + base;
#pragma unroll
            for (int i = 0; i < 5; ++i) {
                int yy = y + i - 2;
                if ((unsigned)yy < (unsigned)H) {
#pragma unroll
                    for (int j = 0; j < 5; ++j) {
                        int xx = x + j - 2;
                        if ((unsigned)xx < (unsigned)W)
                            acc[i * 5 + j] += v2 * src[yy * W + xx];
                    }
                }
            }
        }
#pragma unroll
        for (int k = 0; k < 25; ++k) {
            corr[((size_t)(b * 25 + k)) * HW + r] = acc[k];
            m = fmaxf(m, acc[k]);
        }
    }
    // wave32 reduction then one atomic per wave
#pragma unroll
    for (int off = 16; off > 0; off >>= 1)
        m = fmaxf(m, __shfl_xor(m, off, 32));
    if ((threadIdx.x & 31) == 0) atomicMax(gmax, f2ord(m));
}

// ---------------------------------------------------------------------------
// Softmax over 25 offsets + 25-tap warp:  new[b,c,y,x] = sum_k att_k * f1[shift_k]
// One block handles 64 consecutive pixels; phase 1 builds att in LDS.
// ---------------------------------------------------------------------------
__global__ __launch_bounds__(256)
void softmax_warp_kernel(const float* __restrict__ f1,
                         const float* __restrict__ corr,
                         const unsigned* __restrict__ gmax,
                         float* __restrict__ out,
                         int C, int H, int W) {
    __shared__ float att[64][26];
    int HW = H * W;
    int base = blockIdx.x * 64;
    float scale = 20.f / ord2f(*gmax);
    int tid = threadIdx.x;

    if (tid < 64) {
        int p = base + tid, b = p / HW, r = p % HW;
        float cv[25];
        float m = -1e30f;
#pragma unroll
        for (int k = 0; k < 25; ++k) {
            cv[k] = corr[((size_t)(b * 25 + k)) * HW + r] * scale;
            m = fmaxf(m, cv[k]);
        }
        float s = 0.f;
#pragma unroll
        for (int k = 0; k < 25; ++k) { cv[k] = __expf(cv[k] - m); s += cv[k]; }
        float inv = 1.f / s;
#pragma unroll
        for (int k = 0; k < 25; ++k) att[tid][k] = cv[k] * inv;
    }
    __syncthreads();

    int pp = tid & 63, cofs = tid >> 6;                 // 4 channels in flight
    int p = base + pp, b = p / HW, r = p % HW, y = r / W, x = r % W;
    for (int c = cofs; c < C; c += 4) {
        const float* src = f1 + ((size_t)(b * C + c)) * HW;
        float acc = 0.f;
#pragma unroll
        for (int i = 0; i < 5; ++i) {
            int yy = y + i - 2;
            if ((unsigned)yy < (unsigned)H) {
#pragma unroll
                for (int j = 0; j < 5; ++j) {
                    int xx = x + j - 2;
                    if ((unsigned)xx < (unsigned)W)
                        acc += att[pp][i * 5 + j] * src[yy * W + xx];
                }
            }
        }
        out[((size_t)(b * C + c)) * HW + r] = acc;
    }
}

// ---------------------------------------------------------------------------
// Gating conv: 2C -> 2 channels, 3x3 SAME, sigmoid. (Tiny; WMMA would waste
// 62/64 of the M tile, so direct VALU.)
// ---------------------------------------------------------------------------
__global__ __launch_bounds__(256)
void gate_conv_kernel(const float* __restrict__ key, const float* __restrict__ neww,
                      const float* __restrict__ Wg,   // [2][2C][3][3]
                      const float* __restrict__ bg,   // [2]
                      float* __restrict__ g, int C, int H, int W, int Npix) {
    int p = blockIdx.x * blockDim.x + threadIdx.x;
    if (p >= Npix) return;
    int HW = H * W, b = p / HW, r = p % HW, y = r / W, x = r % W;
    float a0 = bg[0], a1 = bg[1];
    int C2 = 2 * C;
    for (int ic = 0; ic < C2; ++ic) {
        const float* src = (ic < C) ? key + ((size_t)(b * C + ic)) * HW
                                    : neww + ((size_t)(b * C + ic - C)) * HW;
        const float* w0 = Wg + (size_t)ic * 9;
        const float* w1 = Wg + ((size_t)C2 + ic) * 9;
#pragma unroll
        for (int t = 0; t < 9; ++t) {
            int yy = y + t / 3 - 1, xx = x + t % 3 - 1;
            if ((unsigned)yy < (unsigned)H && (unsigned)xx < (unsigned)W) {
                float v = src[yy * W + xx];
                a0 += w0[t] * v;
                a1 += w1[t] * v;
            }
        }
    }
    g[((size_t)(b * 2 + 0)) * HW + r] = 1.f / (1.f + __expf(-a0));
    g[((size_t)(b * 2 + 1)) * HW + r] = 1.f / (1.f + __expf(-a1));
}

// ---------------------------------------------------------------------------
// Implicit-GEMM 3x3 conv via V_WMMA_F32_16X16X32_BF16.
//   M = outC (MUST be a multiple of 64), N = flattened pixels,
//   K = Cin per filter tap (tap-outer iteration, incremental (t,kk) tracking).
//   Block: 128 threads (4 waves), tile 64(M) x 64(N); wave = 32x32 = 2x2 WMMA.
//   Double-buffered LDS; A tile streamed with CDNA5 async global->LDS
//   (global_load_async_to_lds_b128, ASYNCcnt); B tile goes through VGPRs for
//   the fused concat(in0,in1) + per-pixel gating + f32->bf16 convert.
//   Per-tap loader state (validity, spatial offset, base pointers, gate
//   factors) is hoisted: recomputed 9x instead of per k-step. The concat
//   side select is kk < C0 (scalar-uniform: kk, C0 both multiples of 32).
// ---------------------------------------------------------------------------
#define M_BLK 64
#define N_BLK 64
#define K_STEP 32
#define LDSS 40   // padded row stride (bf16 elems) to spread LDS banks

__global__ __launch_bounds__(128)
void conv3x3_wmma_kernel(const float* __restrict__ in0,
                         const float* __restrict__ in1,      // may be null
                         int C0, int Cin,
                         const float* __restrict__ gate,     // may be null [B,2,H,W]
                         const unsigned short* __restrict__ Wb, // bf16 [9][outC][Cin]
                         const float* __restrict__ bias,
                         float* __restrict__ out,
                         int ocBase, int ocStride, int outC,
                         int H, int W, int act) {
    __shared__ unsigned short As[2][M_BLK * LDSS];
    __shared__ unsigned short Bs[2][N_BLK * LDSS];

    const int tid = threadIdx.x;
    const int lane = tid & 31;
    const int wave = tid >> 5;
    const int HW = H * W;
    const int p0 = blockIdx.x * N_BLK;
    const int oc0 = blockIdx.y * M_BLK;

    // B-tile loader coordinates (fixed pixel per thread, 16 channels per pass)
    const int bn  = tid & 63;
    const int bk0 = (tid >> 6) * 16;
    const int pB  = p0 + bn;
    const int bB  = pB / HW;
    const int rB  = pB % HW;
    const int yB  = rB / W;
    const int xB  = rB % W;

    // A-tile loader coordinates: 128 threads x 16B = 64 rows x 32B (bf16 k 0-7, 16-23)
    const int ar  = tid >> 1;
    const int ak0 = (tid & 1) * 16;
    const unsigned short* aRow = Wb + (size_t)(oc0 + ar) * Cin + ak0;
    const size_t aTap = (size_t)outC * Cin;   // A advance per filter tap

    v8f acc[2][2] = {};

    const int wm = (wave >> 1) * 32;
    const int wn = (wave & 1) * 32;
    const int lmod = lane & 15;
    const int lhi  = lane >> 4;

    const int nsteps = 9 * (Cin / K_STEP);

    // ---- per-tap loader state (updated 9x, not per k-step) ----
    bool validL = false;
    const float* b0L = nullptr;   // in0 base at channel bk0, tap-shifted pixel
    const float* b1L = nullptr;   // in1 base at channel bk0, tap-shifted pixel
    float g0L = 1.f, g1L = 1.f;

    auto set_tap = [&](int t) {
        const int dy = t / 3 - 1, dx = t % 3 - 1;
        const int yy = yB + dy, xx = xB + dx;
        validL = ((unsigned)yy < (unsigned)H) && ((unsigned)xx < (unsigned)W);
        const int sp = yy * W + xx;
        b0L = in0 + ((size_t)(bB * C0 + bk0)) * HW + sp;
        if (in1) b1L = in1 + ((size_t)(bB * (Cin - C0) + bk0)) * HW + sp;
        g0L = 1.f; g1L = 1.f;
        if (gate && validL) {
            g0L = gate[((size_t)(bB * 2 + 0)) * HW + sp];
            g1L = gate[((size_t)(bB * 2 + 1)) * HW + sp];
        }
    };

    // ---- load tiles for (tap t, k-offset kk) into LDS buffer `buf` ----
    auto load_tiles = [&](int t, int kk, int buf) {
        // A tile: async global -> LDS, one b128 per lane (EXEC all ones).
        {
            const unsigned short* ga = aRow + (size_t)t * aTap + kk;
            unsigned lofs = (unsigned)(size_t)&As[buf][ar * LDSS + ak0];
            asm volatile("global_load_async_to_lds_b128 %0, %1, off"
                         :: "v"(lofs), "v"(ga)
                         : "memory");
        }
        // B tile: 64 pixels x 32 k, stored [n][k]; gated concat + bf16 cvt.
        {
            unsigned short tmp[16];
            const bool sideA = kk < C0;             // scalar-uniform select
            const float* src = sideA ? b0L : b1L;
            const size_t chof = (size_t)(sideA ? kk : kk - C0) * HW;
            const float gf = sideA ? g0L : g1L;
#pragma unroll
            for (int j = 0; j < 16; ++j) {
                float v = validL ? src[chof + (size_t)j * HW] * gf : 0.f;
                tmp[j] = f32_to_bf16(v);
            }
            *(uint4*)&Bs[buf][bn * LDSS + bk0]     = *(uint4*)&tmp[0];
            *(uint4*)&Bs[buf][bn * LDSS + bk0 + 8] = *(uint4*)&tmp[8];
        }
    };

    // ---- compute one k-step from LDS buffer `buf` ----
    auto compute = [&](int buf) {
        Frag a[2], b[2];
#pragma unroll
        for (int i = 0; i < 2; ++i) {
            int row = wm + i * 16 + lmod;
            int kh = lhi * 8;  // lanes 0-15: k 0-7 / 16-23 ; lanes 16-31: +8
            a[i].q[0] = *(const uint4*)&As[buf][row * LDSS + kh];
            a[i].q[1] = *(const uint4*)&As[buf][row * LDSS + kh + 16];
        }
#pragma unroll
        for (int j = 0; j < 2; ++j) {
            int col = wn + j * 16 + lmod;
            int kb = lhi * 8;  // lanes 0-15: k 0-7 / 16-23 ; lanes 16-31: +8
            b[j].q[0] = *(const uint4*)&Bs[buf][col * LDSS + kb];
            b[j].q[1] = *(const uint4*)&Bs[buf][col * LDSS + kb + 16];
        }
#pragma unroll
        for (int i = 0; i < 2; ++i)
#pragma unroll
            for (int j = 0; j < 2; ++j)
                acc[i][j] = __builtin_amdgcn_wmma_f32_16x16x32_bf16(
                    false, a[i].v, false, b[j].v, (short)0, acc[i][j],
                    false, false);
    };

    // ---- ping-pong main loop: one barrier per k-step ----
    set_tap(0);
    load_tiles(0, 0, 0);
    asm volatile("s_wait_asynccnt 0x0" ::: "memory");
    __syncthreads();

    int t = 0, kk = 0;
    for (int step = 0; step < nsteps; ++step) {
        const int cur = step & 1;
        int kn = kk + K_STEP, tn = t;          // incremental (t,kk), no division
        if (kn == Cin) { kn = 0; tn = t + 1; }
        if (step + 1 < nsteps) {
            if (tn != t) set_tap(tn);
            load_tiles(tn, kn, cur ^ 1);
        }
        compute(cur);
        asm volatile("s_wait_asynccnt 0x0" ::: "memory");
        __syncthreads();
        t = tn; kk = kn;
    }

    // ---- store D: lane L vgpr r -> (M = r + 8*(L/16), N = L%16) ----
#pragma unroll
    for (int i = 0; i < 2; ++i) {
#pragma unroll
        for (int j = 0; j < 2; ++j) {
            int n = wn + j * 16 + lmod;
            int p = p0 + n;
            int b = p / HW, r2 = p % HW;
#pragma unroll
            for (int r = 0; r < 8; ++r) {
                int oc = oc0 + wm + i * 16 + r + 8 * lhi;
                float v = acc[i][j][r] + bias[oc];
                if (act == 1) v = fmaxf(v, 0.f);
                out[((size_t)(b * ocStride + ocBase + oc)) * HW + r2] = v;
            }
        }
    }
}

// ---------------------------------------------------------------------------
// Host orchestration
// ---------------------------------------------------------------------------
extern "C" void kernel_launch(void* const* d_in, const int* in_sizes, int n_in,
                              void* d_out, int out_size, void* d_ws, size_t ws_size,
                              hipStream_t stream) {
    (void)in_sizes; (void)n_in; (void)out_size; (void)ws_size;

    const int Cs[2] = {256, 512};
    const int Hs[2] = {64, 32};
    const int Ws[2] = {64, 32};
    const int B = 4;

    // frames: x{s}{l} at d_in[s*2 + l]
    const float* X[4][2];
    for (int s = 0; s < 4; ++s)
        for (int l = 0; l < 2; ++l)
            X[s][l] = (const float*)d_in[s * 2 + l];

    // workspace arena (reused across levels; sized for level 0, the larger)
    char* ws = (char*)d_ws;
    size_t off = 0;
    auto alloc = [&](size_t bytes) -> void* {
        void* p = ws + off;
        off = (off + bytes + 255) & ~(size_t)255;
        return p;
    };
    const size_t F = (size_t)B * 256 * 64 * 64;      // = 2 * B*512*32*32 elements
    float* newbuf   = (float*)alloc(F * 4);
    float* totalbuf = (float*)alloc(3 * F * 4);
    float* corrbuf  = (float*)alloc((size_t)B * 25 * 64 * 64 * 4);
    float* gbuf     = (float*)alloc((size_t)B * 2 * 64 * 64 * 4);
    unsigned* gmax  = (unsigned*)alloc(256);
    unsigned short* WdBf = (unsigned short*)alloc((size_t)3 * 9 * 512 * 1024 * 2);
    unsigned short* WcBf = (unsigned short*)alloc((size_t)9 * 512 * 1536 * 2);

    float* outp[2] = {(float*)d_out, (float*)d_out + (size_t)B * 256 * 64 * 64};

    for (int l = 0; l < 2; ++l) {
        const int C = Cs[l], H = Hs[l], W = Ws[l];
        const int HW = H * W, Npix = B * HW;
        const float* Wg = (const float*)d_in[8 + 6 * l + 0];
        const float* bg = (const float*)d_in[8 + 6 * l + 1];
        const float* Wd = (const float*)d_in[8 + 6 * l + 2];
        const float* bd = (const float*)d_in[8 + 6 * l + 3];
        const float* Wc = (const float*)d_in[8 + 6 * l + 4];
        const float* bc = (const float*)d_in[8 + 6 * l + 5];

        // --- convert weights to bf16 [tap][outC][Cin] ---
        const size_t nWd = (size_t)9 * C * (2 * C);
        for (int s = 0; s < 3; ++s) {
            size_t n = nWd;
            wconv_kernel<<<(unsigned)((n + 255) / 256), 256, 0, stream>>>(
                Wd + (size_t)s * C * (2 * C) * 9, WdBf + s * nWd, C, 2 * C);
        }
        {
            size_t n = (size_t)9 * C * (3 * C);
            wconv_kernel<<<(unsigned)((n + 255) / 256), 256, 0, stream>>>(
                Wc, WcBf, C, 3 * C);
        }

        for (int s = 0; s < 3; ++s) {
            hipMemsetAsync(gmax, 0, 4, stream);   // ordered-uint identity (< -inf)
            corr_kernel<<<Npix / 256, 256, 0, stream>>>(
                X[3][l], X[s][l], corrbuf, gmax, C, H, W, Npix);
            softmax_warp_kernel<<<Npix / 64, 256, 0, stream>>>(
                X[s][l], corrbuf, gmax, newbuf, C, H, W);
            gate_conv_kernel<<<Npix / 256, 256, 0, stream>>>(
                X[3][l], newbuf, Wg + (size_t)s * 2 * (2 * C) * 9, bg + s * 2,
                gbuf, C, H, W, Npix);
            // gated 2C -> C conv, relu, write into concat slice s of totalbuf
            conv3x3_wmma_kernel<<<dim3(Npix / N_BLK, C / M_BLK), 128, 0, stream>>>(
                X[3][l], newbuf, C, 2 * C, gbuf, WdBf + s * nWd, bd + s * C,
                totalbuf, s * C, 3 * C, C, H, W, /*relu*/ 1);
        }
        // 3C -> C conv, relu, straight to output
        conv3x3_wmma_kernel<<<dim3(Npix / N_BLK, C / M_BLK), 128, 0, stream>>>(
            totalbuf, nullptr, 3 * C, 3 * C, nullptr, WcBf, bc,
            outp[l], 0, C, C, H, W, /*relu*/ 1);
    }
}
